// LSTM_37134287241745
// MI455X (gfx1250) — compile-verified, matrix operand order checked by reading, not run on hardware
//
#include <hip/hip_runtime.h>
#include <hip/hip_bf16.h>
#include <math.h>

// ---------------------------------------------------------------------------
// LSTM forward for MI455X (gfx1250, wave32, WMMA f16 -> f32 accumulate).
//
// Phase 1 (parallel, BW-bound): x_proj = X @ Wx + b4, 68.7 GFLOP but traffic
//   dominated (~400MB -> ~20us @ 23.3 TB/s). Stored f16 in *fragment order*
//   [t][mtile][ntile][lane][8] so phase 2 seeds accumulators with b128 loads.
// Phase 2 (serial): one persistent 1024-thread workgroup, 2048 timesteps.
//   h double-buffered in LDS (2x32KB f16), c in registers, WhT (512KB f16)
//   streamed from L2 every step. One s_barrier per step.
// Phase 3: out = h_T @ W_out^T + b_out in the same kernel tail.
// ---------------------------------------------------------------------------

typedef _Float16 half_t;
typedef __attribute__((ext_vector_type(16))) _Float16 v16h;
typedef __attribute__((ext_vector_type(8)))  _Float16 v8h;
typedef __attribute__((ext_vector_type(8)))  float    v8f;

#define T_STEPS 2048
#define BATCH   64
#define DIM     256
#define GDIM    1024   // 4*H gate columns
#define ODIM    128

// 16-bit A/B fragment: lane L holds row/col (L%16); halves 0..7 = K[kb..kb+7],
// halves 8..15 = K[kb+16..kb+23], kb already includes +8 for lanes >= 16.
// With [N][K] (or [M][K]) f16 storage this is two 16B loads at +0 and +32B.
static __device__ inline v16h load_frag16(const half_t* p) {
  const v8h* q = reinterpret_cast<const v8h*>(p);
  v8h lo = q[0];
  v8h hh = q[2];
  v16h r;
#pragma unroll
  for (int i = 0; i < 8; ++i) { r[i] = lo[i]; r[i + 8] = hh[i]; }
  return r;
}

static __device__ inline v8f wmma_f16(v16h a, v16h b, v8f c) {
  return __builtin_amdgcn_wmma_f32_16x16x32_f16(false, a, false, b,
                                                (short)0, c, false, false);
}

static __device__ inline float sigmoidf_(float x) {
  return 1.0f / (1.0f + __expf(-x));
}

// Fragment-order x_proj index: [t][mtile(4)][ntile(64)][lane(32)][e(8)] f16.
static __device__ inline size_t xp_idx(int t, int mt, int nt, int lane) {
  return ((((size_t)t * 4 + mt) * 64 + nt) * 32 + lane) * 8;
}

// ---------------------------------------------------------------------------
// Kernel 0: pack weights. Gate-major columns g = gate*256 + u.
// ---------------------------------------------------------------------------
__global__ __launch_bounds__(256) void lstm_pack_kernel(
    const float* __restrict__ Wf, const float* __restrict__ bf,
    const float* __restrict__ Wi, const float* __restrict__ bi,
    const float* __restrict__ Wc, const float* __restrict__ bc,
    const float* __restrict__ Wo, const float* __restrict__ bo,
    const float* __restrict__ Wout,
    half_t* __restrict__ WxT, half_t* __restrict__ WhT,
    half_t* __restrict__ WoutT, float* __restrict__ b4) {
  int bid = blockIdx.x, d = threadIdx.x;
  if (bid < GDIM) {
    int gate = bid >> 8, u = bid & 255;
    const float* W = (gate == 0) ? Wf : (gate == 1) ? Wi : (gate == 2) ? Wc : Wo;
    const float* bb = (gate == 0) ? bf : (gate == 1) ? bi : (gate == 2) ? bc : bo;
    WxT[bid * 256 + d] = (half_t)W[u * 512 + d];
    WhT[bid * 256 + d] = (half_t)W[u * 512 + 256 + d];
    if (d == 0) b4[bid] = bb[u];
  } else {
    int r = bid - GDIM;                       // 0..127
    WoutT[r * 256 + d] = (half_t)Wout[r * 256 + d];
  }
}

// ---------------------------------------------------------------------------
// Kernel 1: x_proj GEMM. rows r = t*64+b (131072), cols g (1024), K=256.
// Block = 8 waves; wave w -> Mtile blockIdx.x*8+w, Ntiles blockIdx.y*4..+3.
// 16 consecutive rows share t (16 | 64), so Mtile = (t, mb) with mb = mtile&3,
// matching the recurrence kernel's per-step Mtiles -> store fragment-order.
// ---------------------------------------------------------------------------
__global__ __launch_bounds__(256) void lstm_xproj_kernel(
    const float* __restrict__ X, const half_t* __restrict__ WxT,
    const float* __restrict__ b4, half_t* __restrict__ xp) {
  const int lane  = threadIdx.x & 31;
  const int w     = threadIdx.x >> 5;
  const int laneM = lane & 15;
  const int lhi   = lane >> 4;              // 0 or 1
  const int mtile = blockIdx.x * 8 + w;     // 0..8191
  const int ntb   = blockIdx.y * 4;         // 0..60
  const int t     = mtile >> 2;
  const int mb    = mtile & 3;

  // A row for this lane: r = mtile*16 + laneM -> (b = r&63, t)
  const int b = (mtile * 16 + laneM) & 63;
  const float* Xrow = X + ((size_t)b * T_STEPS + t) * DIM;

  v8f acc[4];
#pragma unroll
  for (int nn = 0; nn < 4; ++nn) acc[nn] = (v8f)(0.0f);

#pragma unroll
  for (int k = 0; k < 8; ++k) {
    const int kb = k * 32 + lhi * 8;
    v16h a;                                  // f32 -> f16 on the fly
#pragma unroll
    for (int i = 0; i < 8; ++i) {
      a[i]     = (half_t)Xrow[kb + i];
      a[i + 8] = (half_t)Xrow[kb + 16 + i];
    }
#pragma unroll
    for (int nn = 0; nn < 4; ++nn) {
      const half_t* Bp = WxT + (size_t)((ntb + nn) * 16 + laneM) * 256 + kb;
      acc[nn] = wmma_f16(a, load_frag16(Bp), acc[nn]);
    }
  }

  // Store in fragment order: one b128 per tile.
#pragma unroll
  for (int nn = 0; nn < 4; ++nn) {
    const float bias = b4[(ntb + nn) * 16 + laneM];
    v8h sto;
#pragma unroll
    for (int e = 0; e < 8; ++e) sto[e] = (half_t)(acc[nn][e] + bias);
    *(v8h*)(xp + xp_idx(t, mb, ntb + nn, lane)) = sto;
  }
}

// ---------------------------------------------------------------------------
// Kernel 2: persistent recurrence, 1 workgroup x 1024 threads (32 waves).
// Wave w: jj = w&15 (16 hidden units), mhalf = w>>4 (2 Mtiles).
// Ntiles {gate*16 + jj} -> all 4 gates of the SAME unit land in the same
// lane / same acc element => no cross-lane exchange; c stays in registers.
// h double-buffered in LDS: read h[t&1], write h[(t+1)&1], 1 barrier/step.
// ---------------------------------------------------------------------------
__global__ __launch_bounds__(1024) void lstm_rec_kernel(
    const half_t* __restrict__ xp, const half_t* __restrict__ WhT,
    const half_t* __restrict__ WoutT, const float* __restrict__ bout,
    float* __restrict__ out) {
  __shared__ half_t hlds[2][BATCH * DIM];   // 2 x 32 KB, h f16 [64][256]

  const int tid = threadIdx.x;
  const int lane = tid & 31;
  const int w = tid >> 5;
  const int laneM = lane & 15;
  const int lhi = lane >> 4;
  const int jj = w & 15;
  const int mhalf = w >> 4;

  // h0 = 0 (buffer 0 only; every step fully rewrites the other buffer)
  for (int i = tid; i < BATCH * DIM; i += 1024) hlds[0][i] = (half_t)0;

  int colg[4];
#pragma unroll
  for (int g = 0; g < 4; ++g) colg[g] = g * 256 + jj * 16 + laneM;
  const int u = jj * 16 + laneM;            // hidden unit, 0..255

  float c[2][8];
#pragma unroll
  for (int m = 0; m < 2; ++m)
#pragma unroll
    for (int e = 0; e < 8; ++e) c[m][e] = 0.0f;

  v8f acc[2][4];
  __syncthreads();

#pragma unroll 1
  for (int t = 0; t < T_STEPS; ++t) {
    const half_t* hcur = hlds[t & 1];
    half_t* hnxt = hlds[(t + 1) & 1];

    // Seed accumulators from fragment-order x_proj: 8 b128 loads.
#pragma unroll
    for (int m = 0; m < 2; ++m)
#pragma unroll
      for (int g = 0; g < 4; ++g) {
        const v8h seed =
            *(const v8h*)(xp + xp_idx(t, 2 * mhalf + m, g * 16 + jj, lane));
#pragma unroll
        for (int e = 0; e < 8; ++e) acc[m][g][e] = (float)seed[e];
      }

    // Prefetch next step's x_proj slab (global_prefetch_b8), 128B/lane.
    if (t + 1 < T_STEPS)
      __builtin_prefetch(
          (const void*)(xp + (size_t)(t + 1) * (BATCH * GDIM) + (tid << 6)),
          0, 3);

    // gates += h @ Wh   (K=256 -> 8 WMMA k-steps; B streams from L2)
#pragma unroll
    for (int k = 0; k < 8; ++k) {
      const int kb = k * 32 + lhi * 8;
      v16h a0 = load_frag16(&hcur[((2 * mhalf + 0) * 16 + laneM) * 256 + kb]);
      v16h a1 = load_frag16(&hcur[((2 * mhalf + 1) * 16 + laneM) * 256 + kb]);
#pragma unroll
      for (int g = 0; g < 4; ++g) {
        v16h bfrag = load_frag16(WhT + (size_t)colg[g] * 256 + kb);
        acc[0][g] = wmma_f16(a0, bfrag, acc[0][g]);
        acc[1][g] = wmma_f16(a1, bfrag, acc[1][g]);
      }
    }

    // Elementwise: f/i/c_hat/o are in-lane; write h_new to the other buffer.
#pragma unroll
    for (int m = 0; m < 2; ++m)
#pragma unroll
      for (int e = 0; e < 8; ++e) {
        const float f  = sigmoidf_(acc[m][0][e]);
        const float ig = sigmoidf_(acc[m][1][e]);
        const float ch = tanhf(acc[m][2][e]);
        const float o  = sigmoidf_(acc[m][3][e]);
        const float cn = f * c[m][e] + ig * ch;
        c[m][e] = cn;
        const float hn = o * tanhf(cn);
        const int row = (2 * mhalf + m) * 16 + e + lhi * 8;
        hnxt[row * 256 + u] = (half_t)hn;
      }

    __syncthreads();   // h_new visible before next step's A-fragment loads
  }

  // ---- out = h_T @ W_out^T + b_out : 4 Mtiles x 8 Ntiles, 1 tile per wave.
  const half_t* hfin = hlds[T_STEPS & 1];   // == hlds[0]
  const int mo = w >> 3;                    // 0..3
  const int no = w & 7;                     // 0..7
  v8f oacc = (v8f)(0.0f);
#pragma unroll
  for (int k = 0; k < 8; ++k) {
    const int kb = k * 32 + lhi * 8;
    v16h a  = load_frag16(&hfin[(mo * 16 + laneM) * 256 + kb]);
    v16h bb = load_frag16(WoutT + (size_t)(no * 16 + laneM) * 256 + kb);
    oacc = wmma_f16(a, bb, oacc);
  }
  const int col = no * 16 + laneM;          // 0..127
  const float bo2 = bout[col];
#pragma unroll
  for (int e = 0; e < 8; ++e) {
    const int row = mo * 16 + e + lhi * 8;  // 0..63
    out[row * ODIM + col] = oacc[e] + bo2;
  }
}

// ---------------------------------------------------------------------------
extern "C" void kernel_launch(void* const* d_in, const int* in_sizes, int n_in,
                              void* d_out, int out_size, void* d_ws, size_t ws_size,
                              hipStream_t stream) {
  const float* X    = (const float*)d_in[0];
  const float* Wf   = (const float*)d_in[1];
  const float* bf   = (const float*)d_in[2];
  const float* Wi   = (const float*)d_in[3];
  const float* bi   = (const float*)d_in[4];
  const float* Wc   = (const float*)d_in[5];
  const float* bc   = (const float*)d_in[6];
  const float* Wo   = (const float*)d_in[7];
  const float* bo   = (const float*)d_in[8];
  const float* Wout = (const float*)d_in[9];
  const float* bout = (const float*)d_in[10];

  const size_t szWxT = (size_t)GDIM * DIM * sizeof(half_t);   // 512 KB
  const size_t szWhT = (size_t)GDIM * DIM * sizeof(half_t);   // 512 KB
  const size_t szWoT = (size_t)ODIM * DIM * sizeof(half_t);   //  64 KB
  const size_t szB4  = (size_t)GDIM * sizeof(float);          //   4 KB
  const size_t szXP  = (size_t)T_STEPS * BATCH * GDIM * sizeof(half_t); // 256 MB
  if (ws_size < szWxT + szWhT + szWoT + szB4 + szXP) return;  // refuse to corrupt

  char* p = (char*)d_ws;
  half_t* WxT   = (half_t*)p; p += szWxT;
  half_t* WhT   = (half_t*)p; p += szWhT;
  half_t* WoutT = (half_t*)p; p += szWoT;
  float*  b4    = (float*)p;  p += szB4;
  half_t* xpbuf = (half_t*)p;

  lstm_pack_kernel<<<GDIM + ODIM, 256, 0, stream>>>(
      Wf, bf, Wi, bi, Wc, bc, Wo, bo, Wout, WxT, WhT, WoutT, b4);

  lstm_xproj_kernel<<<dim3(1024, 16), 256, 0, stream>>>(X, WxT, b4, xpbuf);

  lstm_rec_kernel<<<1, 1024, 0, stream>>>(xpbuf, WhT, WoutT, bout,
                                          (float*)d_out);
}
